// LSTM_7421703487663
// MI455X (gfx1250) — matrix operand order repaired and checked by reading, not compiled, not run on hardware
//
#include <hip/hip_runtime.h>

typedef __attribute__((ext_vector_type(16))) _Float16 v16h;
typedef __attribute__((ext_vector_type(8)))  _Float16 v8h;
typedef __attribute__((ext_vector_type(8)))  float    v8f;

namespace {
constexpr int kH = 20;                 // hidden size
constexpr int kT = 512;                // timesteps
constexpr int kB = 4096;               // batch
constexpr int kWavesPerBlock = 8;
constexpr int kBatchPerWave  = 16;     // one WMMA M-tile per wave
constexpr int kThreads       = kWavesPerBlock * 32;
constexpr int kBatchPerBlock = kWavesPerBlock * kBatchPerWave;
// packed-weight tiles: 8 (layer0) + 8 (layer1 Wih) + 8 (layer1 Whh)
constexpr int kWTiles = 24;
}

#if defined(__has_builtin)
#if __has_builtin(__builtin_amdgcn_tanhf)
#define LSTM_HAVE_TANH 1
#endif
#endif

__device__ __forceinline__ float frcp(float x) { return __builtin_amdgcn_rcpf(x); }

__device__ __forceinline__ float ftanh(float x) {
#ifdef LSTM_HAVE_TANH
    return __builtin_amdgcn_tanhf(x);            // single v_tanh_f32 (CDNA5 TRANS)
#else
    float e = __expf(2.0f * x);                   // v_exp_f32
    return 1.0f - 2.0f * frcp(e + 1.0f);          // v_rcp_f32, no div_scale sequence
#endif
}

__device__ __forceinline__ float fsig(float x) {
#ifdef LSTM_HAVE_TANH
    return fmaf(0.5f, __builtin_amdgcn_tanhf(0.5f * x), 0.5f);
#else
    return frcp(1.0f + __expf(-x));
#endif
}

__global__ __launch_bounds__(kThreads, 1)
void lstm2_wmma_kernel(const float* __restrict__ x,
                       const float* __restrict__ Wih0, const float* __restrict__ Whh0,
                       const float* __restrict__ bih0, const float* __restrict__ bhh0,
                       const float* __restrict__ Wih1, const float* __restrict__ Whh1,
                       const float* __restrict__ bih1, const float* __restrict__ bhh1,
                       const float* __restrict__ fcW,  const float* __restrict__ fcb,
                       float* __restrict__ out)
{
    // Packed B-matrix fragments (K=32 x N=16 tiles), per-lane 16 halves.
    __shared__ __align__(32) _Float16 wlds[kWTiles][32][16];
    // Wave-private h buffers, row-major [M=16][K=32] f16.
    // buf0 = h1 (+ x_t at col 20, 1.0 at col 21) ; buf1 = h2
    __shared__ __align__(16) _Float16 hbuf[kWavesPerBlock][2][16][32];

    const int tid  = threadIdx.x;
    const int wave = tid >> 5;
    const int lane = tid & 31;

    // ---------------- one-time: repack weights into WMMA B-layout ----------------
    // B (32x16) packing used here (contiguous k, matching loadA): lane n holds N = n%16; lanes 0-15 K=0..15,
    // lanes 16-31 K=16..31; half q within lane -> K = q + 16*(n/16).
    for (int idx = tid; idx < kWTiles * 512; idx += kThreads) {
        const int tile = idx >> 9;            // 0..23
        const int r    = idx & 511;
        const int ln   = r >> 4;              // lane 0..31
        const int q    = r & 15;              // half index 0..15
        const int K    = q + 16 * (ln >> 4);  // 0..31
        const int ncol = ((tile & 7) << 4) + (ln & 15);   // 0..127 over 8 N-tiles
        const int mat  = tile >> 3;           // 0 = layer0, 1 = layer1-Wih, 2 = layer1-Whh
        const int gate = ncol >> 5;           // i,f,g,o
        const int u    = ncol & 31;           // hidden unit within gate (pad >= 20)
        float val = 0.0f;
        if (u < kH) {
            const int j = gate * kH + u;      // row of torch weight [4H, *]
            if (mat == 0) {
                if      (K < kH)      val = Whh0[j * kH + K];
                else if (K == kH)     val = Wih0[j];              // D == 1
                else if (K == kH + 1) val = bih0[j] + bhh0[j];
            } else if (mat == 1) {
                if      (K < kH)      val = Wih1[j * kH + K];
                else if (K == kH + 1) val = bih1[j] + bhh1[j];    // bias via h1-buf col21
            } else {
                if      (K < kH)      val = Whh1[j * kH + K];
            }
        }
        wlds[tile][ln][q] = (_Float16)val;
    }
    // zero the h buffers (h0 = c0 = 0, pads = 0)
    for (int idx = tid; idx < kWavesPerBlock * 2 * 16 * 32; idx += kThreads)
        ((_Float16*)hbuf)[idx] = (_Float16)0.0f;
    __syncthreads();

    if (lane < kBatchPerWave)
        hbuf[wave][0][lane][kH + 1] = (_Float16)1.0f;   // constant-1 column for bias

    // hoist packed weights into registers (resident across the whole T loop)
    v16h wb[kWTiles];
#pragma unroll
    for (int i = 0; i < kWTiles; ++i)
        wb[i] = *(const v16h*)(&wlds[i][lane][0]);

    const int bbase = blockIdx.x * kBatchPerBlock + wave * kBatchPerWave;
    const float* xrow = x + (size_t)(bbase + (lane & 15)) * kT;   // D == 1

    // A-matrix fragment addressing: lane n -> M = n%16; lanes 0-15 read K 0..15,
    // lanes 16-31 read K 16..31 (halves packed 2 per VGPR).
    const int aM = lane & 15;
    const int kb = (lane < 16) ? 0 : 16;
    auto loadA = [&](int buf) -> v16h {
        v8h lo = *(const v8h*)(&hbuf[wave][buf][aM][kb]);
        v8h hi = *(const v8h*)(&hbuf[wave][buf][aM][kb + 8]);
        return __builtin_shufflevector(lo, hi, 0,1,2,3,4,5,6,7,8,9,10,11,12,13,14,15);
    };

    // C/D layout: element (VGPR e, lane n) = [M = e + 8*(n/16)][N = n%16 (+16*tile)]
    const int colBase = lane & 15;
    const int mOff    = (lane >> 4) * 8;

    v8f c1[2] = {};   // layer-0 cell state, column halves 0..15 / 16..31
    v8f c2[2] = {};   // layer-1 cell state

    // software-pipelined x: value stored at step t was loaded during step t-1
    _Float16 xh = (_Float16)xrow[0];

    for (int t = 0; t < kT; ++t) {
        float xn = 0.0f;
        if (lane < kBatchPerWave) {
            hbuf[wave][0][lane][kH] = xh;          // x_t into A column 20
            xn = xrow[(t + 1) & (kT - 1)];         // prefetch next x (latency hidden)
        }

        v16h a1 = loadA(0);   // {h1(t-1), x_t, 1, 0...}

        // ---------------- layer 0 ----------------
#pragma unroll
        for (int p = 0; p < 2; ++p) {
            v8f z = {};
            v8f gi = __builtin_amdgcn_wmma_f32_16x16x32_f16(false, a1, false, wb[0 + p], (short)0, z, false, false);
            v8f gf = __builtin_amdgcn_wmma_f32_16x16x32_f16(false, a1, false, wb[2 + p], (short)0, z, false, false);
            v8f gg = __builtin_amdgcn_wmma_f32_16x16x32_f16(false, a1, false, wb[4 + p], (short)0, z, false, false);
            v8f go = __builtin_amdgcn_wmma_f32_16x16x32_f16(false, a1, false, wb[6 + p], (short)0, z, false, false);
            const int col = colBase + 16 * p;
            const bool valid = col < kH;
#pragma unroll
            for (int e = 0; e < 8; ++e) {
                float iv = fsig(gi[e]);
                float fv = fsig(gf[e]);
                float gv = ftanh(gg[e]);
                float ov = fsig(go[e]);
                float cv = fmaf(fv, c1[p][e], iv * gv);
                c1[p][e] = cv;
                float hv = ov * ftanh(cv);
                if (valid) hbuf[wave][0][mOff + e][col] = (_Float16)hv;
            }
        }

        v16h a1n = loadA(0);   // h1(t) (+x, +1)
        v16h a2  = loadA(1);   // h2(t-1)

        // ---------------- layer 1 ----------------
#pragma unroll
        for (int p = 0; p < 2; ++p) {
            v8f z = {};
            v8f gi = __builtin_amdgcn_wmma_f32_16x16x32_f16(false, a2,  false, wb[16 + 0 + p], (short)0, z,  false, false);
            gi     = __builtin_amdgcn_wmma_f32_16x16x32_f16(false, a1n, false, wb[ 8 + 0 + p], (short)0, gi, false, false);
            v8f gf = __builtin_amdgcn_wmma_f32_16x16x32_f16(false, a2,  false, wb[16 + 2 + p], (short)0, z,  false, false);
            gf     = __builtin_amdgcn_wmma_f32_16x16x32_f16(false, a1n, false, wb[ 8 + 2 + p], (short)0, gf, false, false);
            v8f gg = __builtin_amdgcn_wmma_f32_16x16x32_f16(false, a2,  false, wb[16 + 4 + p], (short)0, z,  false, false);
            gg     = __builtin_amdgcn_wmma_f32_16x16x32_f16(false, a1n, false, wb[ 8 + 4 + p], (short)0, gg, false, false);
            v8f go = __builtin_amdgcn_wmma_f32_16x16x32_f16(false, a2,  false, wb[16 + 6 + p], (short)0, z,  false, false);
            go     = __builtin_amdgcn_wmma_f32_16x16x32_f16(false, a1n, false, wb[ 8 + 6 + p], (short)0, go, false, false);
            const int col = colBase + 16 * p;
            const bool valid = col < kH;
#pragma unroll
            for (int e = 0; e < 8; ++e) {
                float iv = fsig(gi[e]);
                float fv = fsig(gf[e]);
                float gv = ftanh(gg[e]);
                float ov = fsig(go[e]);
                float cv = fmaf(fv, c2[p][e], iv * gv);
                c2[p][e] = cv;
                float hv = ov * ftanh(cv);
                if (valid) hbuf[wave][1][mOff + e][col] = (_Float16)hv;
            }
        }

        xh = (_Float16)xn;   // convert prefetched x for next step
    }

    // ---------------- final FC on h2(T-1) ----------------
    if (lane < kBatchPerWave) {
        float acc = fcb[0];
#pragma unroll
        for (int u = 0; u < kH; ++u)
            acc += (float)hbuf[wave][1][lane][u] * fcW[u];
        out[bbase + lane] = acc;
    }
}

extern "C" void kernel_launch(void* const* d_in, const int* in_sizes, int n_in,
                              void* d_out, int out_size, void* d_ws, size_t ws_size,
                              hipStream_t stream) {
    (void)in_sizes; (void)n_in; (void)d_ws; (void)ws_size; (void)out_size;
    const float* x    = (const float*)d_in[0];
    const float* Wih0 = (const float*)d_in[1];
    const float* Whh0 = (const float*)d_in[2];
    const float* bih0 = (const float*)d_in[3];
    const float* bhh0 = (const float*)d_in[4];
    const float* Wih1 = (const float*)d_in[5];
    const float* Whh1 = (const float*)d_in[6];
    const float* bih1 = (const float*)d_in[7];
    const float* bhh1 = (const float*)d_in[8];
    const float* fcW  = (const float*)d_in[9];
    const float* fcb  = (const float*)d_in[10];
    float* out = (float*)d_out;

    dim3 grid(kB / kBatchPerBlock);   // 32 blocks
    dim3 block(kThreads);             // 256 threads = 8 wave32
    hipLaunchKernelGGL(lstm2_wmma_kernel, grid, block, 0, stream,
                       x, Wih0, Whh0, bih0, bhh0, Wih1, Whh1, bih1, bhh1,
                       fcW, fcb, out);
}
